// TritonInt4Linear_30855045054750
// MI455X (gfx1250) — matrix operand order repaired and checked by reading, not compiled, not run on hardware
//
#include <hip/hip_runtime.h>

typedef __attribute__((ext_vector_type(16))) _Float16 v16h;
typedef __attribute__((ext_vector_type(8)))  float    v8f;
typedef __attribute__((ext_vector_type(2)))  _Float16 v2h;
typedef __attribute__((ext_vector_type(4)))  int      v4i;

static constexpr int K_      = 8192;
static constexpr int N_      = 28672;
static constexpr int KHALF   = K_ / 2;       // packed int32 per weight row
static constexpr int GROUP   = 128;
static constexpr int NGROUPS = K_ / GROUP;   // 64
static constexpr int NCHUNK  = K_ / 32;      // 256 K-chunks of 32

// One wave: 16 N-columns x all 64 M rows, full-K reduction.
// Block: 4 waves -> 64 N-columns. Grid: N/64 = 448 blocks.
__global__ __launch_bounds__(128)
void int4_gemm_wmma_kernel(const _Float16* __restrict__ x,
                           const int*      __restrict__ wpk,
                           const _Float16* __restrict__ scales,
                           const _Float16* __restrict__ bias,
                           float*          __restrict__ out)
{
    const int lane  = threadIdx.x & 31;
    const int wave  = threadIdx.x >> 5;
    const int ntile = blockIdx.x * 4 + wave;
    const int col   = ntile * 16 + (lane & 15);   // this lane's N column
    const int khi   = lane >> 4;                  // 0/1: K-half selector

    const int*      wcol = wpk    + (size_t)col * KHALF;
    const _Float16* scol = scales + (size_t)col * NGROUPS;

    // A row base pointers (A layout: both lane halves cover M=0..15, different K)
    const _Float16* xrow[4];
#pragma unroll
    for (int m = 0; m < 4; ++m)
        xrow[m] = x + (size_t)(m * 16 + (lane & 15)) * K_ + khi * 16;

    // ---- fragment loaders -------------------------------------------------
    auto loadA = [&](v16h a[4], int kc) {
        const int kb = kc * 32;
#pragma unroll
        for (int m = 0; m < 4; ++m) {
            union { v16h h; v4i q[2]; } af;
            af.q[0] = *(const v4i*)(xrow[m] + kb);
            af.q[1] = *(const v4i*)(xrow[m] + kb + 8);
            a[m] = af.h;
        }
    };
    auto loadB = [&](v4i b[2], int kc) {          // 8 packed int32 = 16 K values
        const int j0 = kc * 16 + khi * 8;
        b[0] = __builtin_nontemporal_load((const v4i*)(wcol + j0));
        b[1] = __builtin_nontemporal_load((const v4i*)(wcol + j0 + 4));
    };

    // ---- software pipeline prologue --------------------------------------
    v8f  acc[4] = {};
    v4i  braw[4][2];          // depth-4 weight prefetch (HBM latency cover)
    v16h a[2][4];             // depth-1 x ping-pong (L2 latency cover)

#pragma unroll
    for (int c = 0; c < 4; ++c) loadB(braw[c], c);
    loadA(a[0], 0);
    _Float16 s_cur = scol[0];

    // ---- main loop: 64 scale groups x 4 chunks ---------------------------
    for (int g = 0; g < NGROUPS; ++g) {
        const _Float16 s_next = scol[(g + 1 < NGROUPS) ? g + 1 : NGROUPS - 1];
        const v2h      s2  = { s_cur, s_cur };
        const _Float16 off = (_Float16)(-1032.0f) * s_cur;  // (nib+1024)-1032 = nib-8
        const v2h      o2  = { off, off };

        // HBM->L2 prefetch two groups ahead of the register pipeline
        const int gp = (g + 2 < NGROUPS) ? g + 2 : NGROUPS - 1;
        __builtin_prefetch(wcol + gp * (GROUP / 2) + khi * 32, 0, 0);

#pragma unroll
        for (int c = 0; c < 4; ++c) {
            const int kc = g * 4 + c;

            // issue next chunk's A fragments (consumed next chunk)
            loadA(a[(c + 1) & 1], (kc + 1 < NCHUNK) ? kc + 1 : NCHUNK - 1);

            // dequantize current B chunk (loaded 4 chunks ago)
            union { v16h h; v2h p[8]; } bf;
#pragma unroll
            for (int u = 0; u < 2; ++u) {
                const v4i w = braw[c][u];
                const int wv[4] = { w.x, w.y, w.z, w.w };
#pragma unroll
                for (int j = 0; j < 4; ++j) {
                    const unsigned v  = (unsigned)wv[j];
                    const unsigned pk = (v & 0xFu) | ((v & 0xF0u) << 12) | 0x64006400u;
                    bf.p[u * 4 + j] = __builtin_bit_cast(v2h, pk) * s2 + o2;
                }
            }

            // refill this weight buffer 4 chunks ahead
            loadB(braw[c], (kc + 4 < NCHUNK) ? kc + 4 : NCHUNK - 1);

            // 4 WMMAs (one per 16-row M tile) against A loaded last chunk
#pragma unroll
            for (int m = 0; m < 4; ++m)
                acc[m] = __builtin_amdgcn_wmma_f32_16x16x32_f16(
                    /*neg_a=*/false, a[c & 1][m], /*neg_b=*/false, bf.h,
                    /*c_mod=*/(short)0, acc[m], /*reuse_a=*/false, /*reuse_b=*/false);
        }
        s_cur = s_next;
    }

    // ---- epilogue: C/D layout — lane l, VGPR i => M = i + 8*(l>>4), N = l&15
    const float b = (float)bias[col];
#pragma unroll
    for (int m = 0; m < 4; ++m) {
#pragma unroll
        for (int i = 0; i < 8; ++i) {
            const int row = m * 16 + khi * 8 + i;
            out[(size_t)row * N_ + col] = acc[m][i] + b;
        }
    }
}

extern "C" void kernel_launch(void* const* d_in, const int* in_sizes, int n_in,
                              void* d_out, int out_size, void* d_ws, size_t ws_size,
                              hipStream_t stream)
{
    const _Float16* x      = (const _Float16*)d_in[0];
    const int*      wpk    = (const int*)     d_in[1];
    const _Float16* scales = (const _Float16*)d_in[2];
    const _Float16* bias   = (const _Float16*)d_in[3];
    // d_in[4] = group_size scalar (128), baked into kernel constants
    float* out = (float*)d_out;

    dim3 grid(N_ / 64);   // 448 blocks
    dim3 block(128);      // 4 waves, wave32
    int4_gemm_wmma_kernel<<<grid, block, 0, stream>>>(x, wpk, scales, bias, out);
}
